// CausalSelfAttention_87531433492648
// MI455X (gfx1250) — matrix operand order repaired and checked by reading, not compiled, not run on hardware
//
#include <hip/hip_runtime.h>
#include <hip/hip_bf16.h>

// ---------------- problem constants ----------------
#define BATCH  4
#define SEQ    2048
#define DMODEL 384
#define NH     6
#define HD     64        // head dim
#define NROWS  (BATCH * SEQ)       // 8192
#define N_QKV  (3 * DMODEL)        // 1152

typedef __bf16 bf16_t;
typedef bf16_t v16bf __attribute__((ext_vector_type(16)));
typedef bf16_t v8bf  __attribute__((ext_vector_type(8)));
typedef float  v8f   __attribute__((ext_vector_type(8)));
typedef unsigned int u32x4 __attribute__((ext_vector_type(4)));
typedef int i32x8 __attribute__((ext_vector_type(8)));
typedef int i32x4 __attribute__((ext_vector_type(4)));

static __device__ __forceinline__ v16bf cat8(v8bf lo, v8bf hi) {
  return __builtin_shufflevector(lo, hi, 0,1,2,3,4,5,6,7,8,9,10,11,12,13,14,15);
}

static __device__ __forceinline__ v8bf cvt8(v8f x) {
  v8bf r;
#pragma unroll
  for (int i = 0; i < 8; ++i) r[i] = (bf16_t)x[i];
  return r;
}

// ---------------------------------------------------------------------
// Tensor Data Mover: 2D tile load Global->LDS (CDNA5 ISA ch.8 D# layout).
//   dim0/tile0 : contiguous (innermost) extent in elements (2B each)
//   dim1/tile1 : number of rows;  stride0 = row pitch in elements
//   lds_off    : byte offset in workgroup LDS (dynamic-LDS base == 0)
// Group0: [1:0]=count=1, [63:32]=lds_addr, [120:64]=global_addr, [127:126]=2
// Group1: [17:16]=data_size(1->2B), [79:48]=tensor_dim0, [111:80]=tensor_dim1,
//         [127:112]=tile_dim0, [143:128]=tile_dim1, [207:160]=dim0_stride
// This toolchain (clang-23 / therock-10.0) uses the 6-arg builtin:
//   (u32x4 g0, i32x8 g1, i32x4 g2, i32x4 g3, i32x8 g4, i32 cpol)
// ---------------------------------------------------------------------
static __device__ __forceinline__ void tdm_load_2d(const void* gptr, unsigned lds_off,
                                                   unsigned dim0, unsigned dim1,
                                                   unsigned stride0)
{
  const unsigned long long ga = (unsigned long long)gptr;
  u32x4 g0;
  g0[0] = 1u;                                         // count=1, user mode
  g0[1] = lds_off;                                    // LDS byte address
  g0[2] = (unsigned)(ga & 0xFFFFFFFFu);               // global_addr[31:0]
  g0[3] = (unsigned)((ga >> 32) & 0x01FFFFFFu) | 0x80000000u; // [56:32] | type=2
  i32x8 g1;
  g1[0] = (int)(1u << 16);                            // data_size=1 (2 bytes)
  g1[1] = (int)((dim0 & 0xFFFFu) << 16);              // tensor_dim0 lo16
  g1[2] = (int)(((dim0 >> 16) & 0xFFFFu) | ((dim1 & 0xFFFFu) << 16));
  g1[3] = (int)(((dim1 >> 16) & 0xFFFFu) | ((dim0 & 0xFFFFu) << 16)); // tile_dim0
  g1[4] = (int)(dim1 & 0xFFFFu);                      // tile_dim1 (tile_dim2=0)
  g1[5] = (int)stride0;                               // tensor_dim0_stride lo32
  g1[6] = 0;
  g1[7] = 0;
  i32x4 z4 = {0, 0, 0, 0};                            // 2D: groups 2/3 unused
  i32x8 z8 = {0, 0, 0, 0, 0, 0, 0, 0};
  __builtin_amdgcn_tensor_load_to_lds(g0, g1, z4, z4, z8, 0);
}

// =====================================================================
// Kernel 1: QKV projection  qkv = x @ W^T + b   (M=8192, K=384, N=1152)
// Writes q,k as bf16 [B,H,T,HD] and v TRANSPOSED as bf16 [B,H,HD,T].
// =====================================================================
__global__ __launch_bounds__(256)
void qkv_wmma_kernel(const float* __restrict__ x, const float* __restrict__ w,
                     const float* __restrict__ bias,
                     bf16_t* __restrict__ qw, bf16_t* __restrict__ kw,
                     bf16_t* __restrict__ vtw)
{
  const int lane = threadIdx.x & 31;
  const int wave = threadIdx.x >> 5;
  const int half = lane >> 4;
  const int lm   = lane & 15;
  const int rowBase = blockIdx.x * 128 + wave * 16;
  const int colBase = blockIdx.y * 64;

  v8f acc[4] = {};
  const float* arow = x + (size_t)(rowBase + lm) * DMODEL;

  for (int k0 = 0; k0 < DMODEL; k0 += 32) {
    // A fragment 16x32 bf16 (lanes 0-15: K {0..7,16..23}; 16-31: {8..15,24..31})
    const int aoff = k0 + (half ? 8 : 0);
    if (k0 + 32 < DMODEL) __builtin_prefetch(arow + aoff + 32, 0, 1);
    v16bf afrag = cat8(cvt8(*(const v8f*)(arow + aoff)),
                       cvt8(*(const v8f*)(arow + aoff + 16)));
#pragma unroll
    for (int nt = 0; nt < 4; ++nt) {
      const float* brow = w + (size_t)(colBase + nt * 16 + lm) * DMODEL
                            + k0 + half * 8;
      v16bf bfrag = cat8(cvt8(*(const v8f*)brow),
                         cvt8(*(const v8f*)(brow + 16)));
      acc[nt] = __builtin_amdgcn_wmma_f32_16x16x32_bf16(
          false, afrag, false, bfrag, (short)0, acc[nt], false, false);
    }
  }

#pragma unroll
  for (int nt = 0; nt < 4; ++nt) {
    const int n = colBase + nt * 16 + lm;
    const float bv = bias[n];
    const int sec    = n / DMODEL;
    const int within = n % DMODEL;
    const int h = within / HD;
    const int d = within % HD;
#pragma unroll
    for (int r = 0; r < 8; ++r) {
      const int m = rowBase + r + 8 * half;
      const int b = m / SEQ, t = m % SEQ;
      const bf16_t v = (bf16_t)(acc[nt][r] + bv);
      const size_t head = (size_t)(b * NH + h);
      if (sec == 0)      qw [(head * SEQ + t) * HD + d] = v;
      else if (sec == 1) kw [(head * SEQ + t) * HD + d] = v;
      else               vtw[(head * HD  + d) * SEQ + t] = v;
    }
  }
}

// =====================================================================
// Kernel 2: causal flash attention, one wave per 16-query tile.
// K/V tiles staged Global->LDS by the Tensor Data Mover, double-buffered
// (s_wait_tensorcnt 2 overlaps next-block DMA with current WMMA work).
// Dynamic LDS layout (bytes):
//   [0,8192)      sK[2][32][64] bf16
//   [8192,16384)  sV[2][64][32] bf16   (V transposed: rows are head-dims)
//   [16384,18496) sP[16][33]   f32    (P C-layout -> A-layout staging)
// =====================================================================
#define LDS_K0   0u
#define LDS_V0   8192u
#define LDS_P    16384u
#define LDS_SZ   18496u

__global__ __launch_bounds__(32)
void flash_attn_kernel(const bf16_t* __restrict__ qw, const bf16_t* __restrict__ kw,
                       const bf16_t* __restrict__ vtw, bf16_t* __restrict__ ow)
{
  extern __shared__ char smem[];
  bf16_t* sK = (bf16_t*)(smem + LDS_K0);   // [2][32*64]
  bf16_t* sV = (bf16_t*)(smem + LDS_V0);   // [2][64*32]
  float*  sP = (float*) (smem + LDS_P);    // [16*33]

  const int lane = threadIdx.x & 31;
  const int half = lane >> 4;
  const int lm   = lane & 15;
  const int qtile = blockIdx.x;
  const int bh    = blockIdx.y;            // b*NH + h
  const int b = bh / NH, h = bh % NH;

  // Q A-fragments for the two 32-wide d-chunks of HD=64
  const bf16_t* qrow = qw + (((size_t)bh * SEQ) + qtile * 16 + lm) * HD;
  v16bf qa[2];
#pragma unroll
  for (int dc = 0; dc < 2; ++dc) {
    const int off = dc * 32 + (half ? 8 : 0);
    qa[dc] = cat8(*(const v8bf*)(qrow + off), *(const v8bf*)(qrow + off + 16));
  }

  v8f acc[4] = {};
  float mrow[8], lrow[8];
#pragma unroll
  for (int r = 0; r < 8; ++r) { mrow[r] = -1e30f; lrow[r] = 0.f; }

  const int kmax    = (qtile + 1) * 16;
  const int nblocks = (kmax + 31) / 32;
  const float scale = 0.125f;              // 1/sqrt(64)

  // TDM prefetch of key-block 0 into buffer 0
  const bf16_t* kbase = kw  + (size_t)bh * SEQ * HD;
  const bf16_t* vbase = vtw + (size_t)bh * HD * SEQ;
  tdm_load_2d(kbase, LDS_K0, HD, 32, HD);                    // K: 32 rows x 64
  tdm_load_2d(vbase, LDS_V0, 32, HD, SEQ);                   // V^T: 64 rows x 32

  int buf = 0;
  for (int kb = 0; kb < nblocks; ++kb) {
    const int t0 = kb * 32;

    // issue next block's DMA into the other buffer, then wait for current
    if (kb + 1 < nblocks) {
      const int t1 = t0 + 32;
      tdm_load_2d(kbase + (size_t)t1 * HD, LDS_K0 + (buf ^ 1) * 4096u, HD, 32, HD);
      tdm_load_2d(vbase + t1,              LDS_V0 + (buf ^ 1) * 4096u, 32, HD, SEQ);
      __builtin_amdgcn_s_wait_tensorcnt(2);   // current buffer complete
    } else {
      __builtin_amdgcn_s_wait_tensorcnt(0);   // drain on last block
    }
    asm volatile("" ::: "memory");

    const bf16_t* kt = sK + buf * (32 * 64);
    const bf16_t* vt = sV + buf * (64 * 32);

    // ---- S = Q @ K^T (two 16-wide key tiles) from LDS ----
    v8f s[2] = {};
#pragma unroll
    for (int nt = 0; nt < 2; ++nt) {
      const bf16_t* krow = kt + (nt * 16 + lm) * HD + half * 8;
#pragma unroll
      for (int dc = 0; dc < 2; ++dc) {
        v16bf kb16 = cat8(*(const v8bf*)(krow + dc * 32),
                          *(const v8bf*)(krow + dc * 32 + 16));
        s[nt] = __builtin_amdgcn_wmma_f32_16x16x32_bf16(
            false, qa[dc], false, kb16, (short)0, s[nt], false, false);
      }
    }

    // ---- scale + causal mask (C layout: elem r -> row r+8*half, col lm) ----
    const int qr = qtile * 16 + 8 * half;
#pragma unroll
    for (int nt = 0; nt < 2; ++nt) {
      const int kc = t0 + nt * 16 + lm;
#pragma unroll
      for (int r = 0; r < 8; ++r)
        s[nt][r] = (kc <= qr + r) ? s[nt][r] * scale : -1e30f;
    }

    // ---- row max across 32 cols ----
    float rmax[8];
#pragma unroll
    for (int r = 0; r < 8; ++r) rmax[r] = fmaxf(s[0][r], s[1][r]);
#pragma unroll
    for (int msk = 1; msk < 16; msk <<= 1)
#pragma unroll
      for (int r = 0; r < 8; ++r)
        rmax[r] = fmaxf(rmax[r], __shfl_xor(rmax[r], msk, 32));

    // ---- online softmax update ----
    float alpha[8], p0[8], p1[8], rsum[8];
#pragma unroll
    for (int r = 0; r < 8; ++r) {
      const float mnew = fmaxf(mrow[r], rmax[r]);
      alpha[r] = __expf(mrow[r] - mnew);
      p0[r] = __expf(s[0][r] - mnew);
      p1[r] = __expf(s[1][r] - mnew);
      rsum[r] = p0[r] + p1[r];
      mrow[r] = mnew;
    }
#pragma unroll
    for (int msk = 1; msk < 16; msk <<= 1)
#pragma unroll
      for (int r = 0; r < 8; ++r)
        rsum[r] += __shfl_xor(rsum[r], msk, 32);
#pragma unroll
    for (int r = 0; r < 8; ++r) {
      lrow[r] = lrow[r] * alpha[r] + rsum[r];
#pragma unroll
      for (int nt = 0; nt < 4; ++nt) acc[nt][r] *= alpha[r];
    }

    // ---- P: C-layout -> A-layout via LDS ----
    __syncthreads();
#pragma unroll
    for (int r = 0; r < 8; ++r) {
      sP[(r + 8 * half) * 33 + lm]      = p0[r];
      sP[(r + 8 * half) * 33 + 16 + lm] = p1[r];
    }
    __syncthreads();
    v16bf pa;
#pragma unroll
    for (int i = 0; i < 8; ++i) {
      pa[i]     = (bf16_t)sP[lm * 33 + half * 8 + i];
      pa[i + 8] = (bf16_t)sP[lm * 33 + 16 + half * 8 + i];
    }

    // ---- O += P @ V from LDS (V already transposed) ----
#pragma unroll
    for (int nt = 0; nt < 4; ++nt) {
      const bf16_t* vrow = vt + (nt * 16 + lm) * 32 + half * 8;
      v16bf vb = cat8(*(const v8bf*)(vrow), *(const v8bf*)(vrow + 16));
      acc[nt] = __builtin_amdgcn_wmma_f32_16x16x32_bf16(
          false, pa, false, vb, (short)0, acc[nt], false, false);
    }

    buf ^= 1;
  }

  // ---- normalize and store attention output as bf16 [B,T,D] ----
#pragma unroll
  for (int nt = 0; nt < 4; ++nt) {
#pragma unroll
    for (int r = 0; r < 8; ++r) {
      const int t = qtile * 16 + r + 8 * half;
      const int dcol = h * HD + nt * 16 + lm;
      ow[((size_t)b * SEQ + t) * DMODEL + dcol] = (bf16_t)(acc[nt][r] / lrow[r]);
    }
  }
}

// =====================================================================
// Kernel 3: output projection  out = attn @ W_out^T + b   (f32 output)
// =====================================================================
__global__ __launch_bounds__(256)
void out_proj_kernel(const bf16_t* __restrict__ a, const float* __restrict__ w,
                     const float* __restrict__ bias, float* __restrict__ out)
{
  const int lane = threadIdx.x & 31;
  const int wave = threadIdx.x >> 5;
  const int half = lane >> 4;
  const int lm   = lane & 15;
  const int rowBase = blockIdx.x * 128 + wave * 16;
  const int colBase = blockIdx.y * 64;

  v8f acc[4] = {};
  const bf16_t* arow = a + (size_t)(rowBase + lm) * DMODEL;

  for (int k0 = 0; k0 < DMODEL; k0 += 32) {
    const int aoff = k0 + (half ? 8 : 0);
    if (k0 + 32 < DMODEL) __builtin_prefetch(arow + aoff + 32, 0, 1);
    v16bf afrag = cat8(*(const v8bf*)(arow + aoff),
                       *(const v8bf*)(arow + aoff + 16));
#pragma unroll
    for (int nt = 0; nt < 4; ++nt) {
      const float* brow = w + (size_t)(colBase + nt * 16 + lm) * DMODEL
                            + k0 + half * 8;
      v16bf bfrag = cat8(cvt8(*(const v8f*)brow),
                         cvt8(*(const v8f*)(brow + 16)));
      acc[nt] = __builtin_amdgcn_wmma_f32_16x16x32_bf16(
          false, afrag, false, bfrag, (short)0, acc[nt], false, false);
    }
  }

#pragma unroll
  for (int nt = 0; nt < 4; ++nt) {
    const int n = colBase + nt * 16 + lm;
    const float bv = bias[n];
#pragma unroll
    for (int r = 0; r < 8; ++r) {
      const int m = rowBase + r + 8 * half;
      out[(size_t)m * DMODEL + n] = acc[nt][r] + bv;
    }
  }
}

// =====================================================================
extern "C" void kernel_launch(void* const* d_in, const int* in_sizes, int n_in,
                              void* d_out, int out_size, void* d_ws, size_t ws_size,
                              hipStream_t stream) {
  const float* x     = (const float*)d_in[0];
  const float* qkv_w = (const float*)d_in[1];
  const float* qkv_b = (const float*)d_in[2];
  const float* out_w = (const float*)d_in[3];
  const float* out_b = (const float*)d_in[4];
  float* out = (float*)d_out;

  // bf16 staging in workspace: q, k, v^T, attn-out  (4 x 6 MB = 24 MB)
  char* ws = (char*)d_ws;
  const size_t per = (size_t)NROWS * DMODEL * sizeof(bf16_t);
  bf16_t* qws  = (bf16_t*)(ws);
  bf16_t* kws  = (bf16_t*)(ws + per);
  bf16_t* vtws = (bf16_t*)(ws + 2 * per);
  bf16_t* aws  = (bf16_t*)(ws + 3 * per);

  dim3 g1(NROWS / 128, N_QKV / 64);          // 64 x 18
  qkv_wmma_kernel<<<g1, 256, 0, stream>>>(x, qkv_w, qkv_b, qws, kws, vtws);

  dim3 g2(SEQ / 16, BATCH * NH);             // 128 x 24, one wave per block
  flash_attn_kernel<<<g2, 32, LDS_SZ, stream>>>(qws, kws, vtws, aws);

  dim3 g3(NROWS / 128, DMODEL / 64);         // 64 x 6
  out_proj_kernel<<<g3, 256, 0, stream>>>(aws, out_w, out_b, out);
}
